// Attention_52355651338291
// MI455X (gfx1250) — matrix operand order repaired and checked by reading, not compiled, not run on hardware
//
#include <hip/hip_runtime.h>

// ---------------------------------------------------------------------------
// MI455X (gfx1250) attention: fp32 inputs -> bf16 WMMA (f32 accum) flash attn.
// All data (~50MB) fits in 192MB L2 -> compute bound on v_wmma_f32_16x16x32_bf16.
// Layout choices make every WMMA A/B fragment a contiguous 16/32B load per lane.
// ---------------------------------------------------------------------------

typedef __bf16 bf16;
typedef __attribute__((ext_vector_type(16))) __bf16 v16bf;
typedef __attribute__((ext_vector_type(8)))  __bf16 v8bf;
typedef __attribute__((ext_vector_type(8)))  float  v8f;

#define HEADS 8
#define DHEAD 64
#define BATCH 4
#define DIM   256
#define LSEQ  2048
#define HID   512            // HEADS*DHEAD
#define SCALE_LOG2E 0.180336878f   // (1/sqrt(64)) * log2(e) -> softmax via exp2

__device__ __forceinline__ v8f wmma_bf16(v16bf a, v16bf b, v8f c) {
  // D = A(16x32 bf16) * B(32x16 bf16) + C(16x16 f32)
  return __builtin_amdgcn_wmma_f32_16x16x32_bf16(
      false, a, false, b, (short)0, c, false, false);
}

// A-fragment (16-bit, 16x32): lane m=lane&15; K-run [base..base+15]
// where base = kk*32 + (lane>>4)*16.  One 32B load.
__device__ __forceinline__ v16bf load_afrag(const bf16* p) {
  return *(const v16bf*)p;
}

__device__ __forceinline__ v8f vzero8() {
  v8f z = {0.f, 0.f, 0.f, 0.f, 0.f, 0.f, 0.f, 0.f};
  return z;
}

// ------------------------------ prep kernels -------------------------------
__global__ void prep_x(const float* __restrict__ x, bf16* __restrict__ xT) {
  int i = blockIdx.x * 256 + threadIdx.x;              // over b*c*l
  if (i >= BATCH * DIM * LSEQ) return;
  int b = i / (DIM * LSEQ);
  int r = i - b * DIM * LSEQ;
  int c = r / LSEQ;
  int l = r - c * LSEQ;
  xT[((size_t)b * LSEQ + l) * DIM + c] = (bf16)x[i];   // xT: [b, l, c] bf16
}

__global__ void prep_wqkv(const float* __restrict__ w, bf16* __restrict__ wb) {
  int i = blockIdx.x * 256 + threadIdx.x;              // over 1536*256
  if (i >= 3 * HID * DIM) return;
  int o = i >> 8;                                      // row (DIM=256 cols)
  float s = (o < HID) ? SCALE_LOG2E : 1.0f;            // fold softmax scale into Q
  wb[i] = (bf16)(w[i] * s);
}

__global__ void prep_wout(const float* __restrict__ w, bf16* __restrict__ wb) {
  int i = blockIdx.x * 256 + threadIdx.x;              // over 256*512
  if (i >= DIM * HID) return;
  wb[i] = (bf16)w[i];
}

// ------------------------------ QKV projection -----------------------------
// Computes qkv^T[l,o] = xT[l,:] . w_qkv[o,:]  (K = 256), per batch.
// Wave: 16(M=l) x 64(N=o) tile; block = 4 waves -> 64x64.
// Epilogue scatters into Q,K [b,h,l,d] and V [b,h,d,l] (bf16).
__global__ __launch_bounds__(128) void qkv_gemm(
    const bf16* __restrict__ xT, const bf16* __restrict__ wb,
    bf16* __restrict__ Qb, bf16* __restrict__ Kb, bf16* __restrict__ Vb) {
  const int lane = threadIdx.x & 31;
  const int wave = threadIdx.x >> 5;
  const int half = lane >> 4;
  const int l15  = lane & 15;
  const int b     = blockIdx.z;
  const int mbase = blockIdx.y * 64 + wave * 16;       // l
  const int nbase = blockIdx.x * 64;                   // o

  const bf16* arow = xT + ((size_t)b * LSEQ + mbase + l15) * DIM;
  v8f acc[4] = {vzero8(), vzero8(), vzero8(), vzero8()};

  for (int kk = 0; kk < DIM; kk += 32) {
    v16bf a = load_afrag(arow + kk + half * 16);
#pragma unroll
    for (int nt = 0; nt < 4; ++nt) {
      const bf16* bp = wb + (size_t)(nbase + nt * 16 + l15) * DIM + kk + half * 16;
      v16bf bv = *(const v16bf*)bp;                    // 32B contiguous
      acc[nt] = wmma_bf16(a, bv, acc[nt]);
    }
  }
#pragma unroll
  for (int nt = 0; nt < 4; ++nt) {
#pragma unroll
    for (int r = 0; r < 8; ++r) {
      int o  = nbase + nt * 16 + l15;
      int li = mbase + r + 8 * half;
      bf16 v = (bf16)acc[nt][r];
      if (o < HID) {                                   // Q: [b,h,l,d]
        int h = o >> 6, d = o & 63;
        Qb[(((size_t)b * HEADS + h) * LSEQ + li) * DHEAD + d] = v;
      } else if (o < 2 * HID) {                        // K: [b,h,l,d]
        int oo = o - HID; int h = oo >> 6, d = oo & 63;
        Kb[(((size_t)b * HEADS + h) * LSEQ + li) * DHEAD + d] = v;
      } else {                                         // V: [b,h,d,l]
        int oo = o - 2 * HID; int h = oo >> 6, d = oo & 63;
        Vb[(((size_t)b * HEADS + h) * DHEAD + d) * LSEQ + li] = v;
      }
    }
  }
}

// ------------------------------ flash attention ----------------------------
// Per wave: 16 query rows, stream 64 keys/step with online softmax (exp2; the
// 1/sqrt(d)*log2e factor is folded into Q). S = Q.K^T via WMMA (K rows give
// contiguous B frags); P staged through LDS (C-layout -> A-layout); O += P.V
// with V [d,l] giving contiguous B frags. Output: O2 [b, l, hid] bf16.
__global__ __launch_bounds__(128) void attn_kernel(
    const bf16* __restrict__ Q, const bf16* __restrict__ K,
    const bf16* __restrict__ V, bf16* __restrict__ O2) {
  __shared__ __align__(128) bf16 ldsP[4][16 * 64];     // 2KB per wave
  const int lane = threadIdx.x & 31;
  const int wave = threadIdx.x >> 5;
  const int half = lane >> 4;
  const int l15  = lane & 15;
  const int bh   = blockIdx.y;                          // b*HEADS + h
  const int b    = bh >> 3;
  const int h    = bh & 7;
  const int mbase = blockIdx.x * 64 + wave * 16;        // query row base

  const bf16* Qbh = Q + (size_t)bh * LSEQ * DHEAD;      // [l,d]
  const bf16* Kbh = K + (size_t)bh * LSEQ * DHEAD;      // [l,d]
  const bf16* Vbh = V + (size_t)bh * DHEAD * LSEQ;      // [d,l]

  // Q A-fragments (held for the whole pass)
  v16bf qa[2];
#pragma unroll
  for (int kk = 0; kk < 2; ++kk)
    qa[kk] = load_afrag(Qbh + (size_t)(mbase + l15) * DHEAD + kk * 32 + half * 16);

  v8f oacc[4] = {vzero8(), vzero8(), vzero8(), vzero8()};
  float mrow[8], lrow[8];
#pragma unroll
  for (int r = 0; r < 8; ++r) { mrow[r] = -__builtin_huge_valf(); lrow[r] = 0.f; }

  for (int jb = 0; jb < LSEQ; jb += 64) {
    // ---- S = Q . K^T  (16 x 64 tile) ----
    v8f s[4] = {vzero8(), vzero8(), vzero8(), vzero8()};
#pragma unroll
    for (int kk = 0; kk < 2; ++kk) {
#pragma unroll
      for (int nt = 0; nt < 4; ++nt) {
        const bf16* kp = Kbh + (size_t)(jb + nt * 16 + l15) * DHEAD + kk * 32 + half * 16;
        v16bf kb = *(const v16bf*)kp;
        s[nt] = wmma_bf16(qa[kk], kb, s[nt]);
      }
    }
    // ---- online softmax (rows live on 16-lane halves; reduce with shfl_xor) ----
    float alpha[8];
#pragma unroll
    for (int r = 0; r < 8; ++r) {
      float bm = fmaxf(fmaxf(s[0][r], s[1][r]), fmaxf(s[2][r], s[3][r]));
#pragma unroll
      for (int m = 1; m < 16; m <<= 1) bm = fmaxf(bm, __shfl_xor(bm, m, 32));
      float mnew = fmaxf(mrow[r], bm);
      alpha[r] = exp2f(mrow[r] - mnew);                // exp2f(-inf)=0 on first step
      mrow[r] = mnew;
    }
    float psum[8];
#pragma unroll
    for (int r = 0; r < 8; ++r) psum[r] = 0.f;
#pragma unroll
    for (int nt = 0; nt < 4; ++nt)
#pragma unroll
      for (int r = 0; r < 8; ++r) {
        float p = exp2f(s[nt][r] - mrow[r]);
        s[nt][r] = p;
        psum[r] += p;
      }
#pragma unroll
    for (int r = 0; r < 8; ++r) {
      float ps = psum[r];
#pragma unroll
      for (int m = 1; m < 16; m <<= 1) ps += __shfl_xor(ps, m, 32);
      lrow[r] = lrow[r] * alpha[r] + ps;
    }
#pragma unroll
    for (int dt = 0; dt < 4; ++dt)
#pragma unroll
      for (int r = 0; r < 8; ++r) oacc[dt][r] *= alpha[r];

    // ---- stage P through LDS: C-layout -> A-layout ----
    __syncthreads();                                   // WAR on previous reads
#pragma unroll
    for (int nt = 0; nt < 4; ++nt)
#pragma unroll
      for (int r = 0; r < 8; ++r)
        ldsP[wave][(r + 8 * half) * 64 + nt * 16 + l15] = (bf16)s[nt][r];
    __syncthreads();                                   // RAW before frag reads
    v16bf pa[2];
#pragma unroll
    for (int kk = 0; kk < 2; ++kk)
      pa[kk] = load_afrag(&ldsP[wave][l15 * 64 + kk * 32 + half * 16]);

    // ---- O += P . V ----
#pragma unroll
    for (int kk = 0; kk < 2; ++kk) {
#pragma unroll
      for (int dt = 0; dt < 4; ++dt) {
        const bf16* vp = Vbh + (size_t)(dt * 16 + l15) * LSEQ + jb + kk * 32 + half * 16;
        v16bf vb = *(const v16bf*)vp;
        oacc[dt] = wmma_bf16(pa[kk], vb, oacc[dt]);
      }
    }
  }

  // ---- normalize and store: O2 [b, l, hid] ----
#pragma unroll
  for (int r = 0; r < 8; ++r) {
    float inv = 1.0f / lrow[r];
    int li = mbase + r + 8 * half;
#pragma unroll
    for (int dt = 0; dt < 4; ++dt) {
      int d = dt * 16 + l15;
      O2[((size_t)b * LSEQ + li) * HID + h * DHEAD + d] = (bf16)(oacc[dt][r] * inv);
    }
  }
}

// ------------------------------ output projection --------------------------
// final^T[l,o] = O2[l,:] . w_out[o,:]  (K = 512), + bias; store f32 [b,256,l].
__global__ __launch_bounds__(128) void out_gemm(
    const bf16* __restrict__ O2, const bf16* __restrict__ wob,
    const float* __restrict__ bias, float* __restrict__ out) {
  const int lane = threadIdx.x & 31;
  const int wave = threadIdx.x >> 5;
  const int half = lane >> 4;
  const int l15  = lane & 15;
  const int b     = blockIdx.z;
  const int mbase = blockIdx.y * 64 + wave * 16;       // l
  const int nbase = blockIdx.x * 64;                   // o

  const bf16* arow = O2 + ((size_t)b * LSEQ + mbase + l15) * HID;
  v8f acc[4] = {vzero8(), vzero8(), vzero8(), vzero8()};

  for (int kk = 0; kk < HID; kk += 32) {
    v16bf a = load_afrag(arow + kk + half * 16);
#pragma unroll
    for (int nt = 0; nt < 4; ++nt) {
      const bf16* bp = wob + (size_t)(nbase + nt * 16 + l15) * HID + kk + half * 16;
      v16bf bv = *(const v16bf*)bp;
      acc[nt] = wmma_bf16(a, bv, acc[nt]);
    }
  }
#pragma unroll
  for (int nt = 0; nt < 4; ++nt) {
    int o = nbase + nt * 16 + l15;
    float bo = bias[o];
#pragma unroll
    for (int r = 0; r < 8; ++r) {
      int li = mbase + r + 8 * half;
      out[((size_t)b * DIM + o) * LSEQ + li] = acc[nt][r] + bo;
    }
  }
}

// ------------------------------ launcher -----------------------------------
extern "C" void kernel_launch(void* const* d_in, const int* in_sizes, int n_in,
                              void* d_out, int out_size, void* d_ws, size_t ws_size,
                              hipStream_t stream) {
  const float* x     = (const float*)d_in[0];   // [4,256,2048]
  const float* w_qkv = (const float*)d_in[1];   // [1536,256]
  const float* w_out = (const float*)d_in[2];   // [256,512]
  const float* b_out = (const float*)d_in[3];   // [256]
  float* out = (float*)d_out;                   // [4,256,2048]

  // workspace layout (bf16 elements, all sections 32B-aligned): ~38.8 MB
  bf16* xT    = (bf16*)d_ws;                              // 4*2048*256
  bf16* wqkvb = xT    + (size_t)BATCH * LSEQ * DIM;       // 1536*256
  bf16* woutb = wqkvb + (size_t)3 * HID * DIM;            // 256*512
  bf16* Qb    = woutb + (size_t)DIM * HID;                // 4*8*2048*64
  bf16* Kb    = Qb    + (size_t)BATCH * HEADS * LSEQ * DHEAD;
  bf16* Vb    = Kb    + (size_t)BATCH * HEADS * LSEQ * DHEAD;
  bf16* O2    = Vb    + (size_t)BATCH * HEADS * LSEQ * DHEAD;  // 4*2048*512

  prep_x   <<<(BATCH * DIM * LSEQ + 255) / 256, 256, 0, stream>>>(x, xT);
  prep_wqkv<<<(3 * HID * DIM + 255) / 256,      256, 0, stream>>>(w_qkv, wqkvb);
  prep_wout<<<(DIM * HID + 255) / 256,          256, 0, stream>>>(w_out, woutb);

  qkv_gemm   <<<dim3(24, 32, BATCH), 128, 0, stream>>>(xT, wqkvb, Qb, Kb, Vb);
  attn_kernel<<<dim3(LSEQ / 64, BATCH * HEADS), 128, 0, stream>>>(Qb, Kb, Vb, O2);
  out_gemm   <<<dim3(4, 32, BATCH), 128, 0, stream>>>(O2, woutb, b_out, out);
}
